// WindowAttention_16449724745596
// MI455X (gfx1250) — matrix operand order repaired and checked by reading, not compiled, not run on hardware
//
#include <hip/hip_runtime.h>

// -------------------------------------------------------------------------
// Fused window attention for MI455X (gfx1250, wave32, WMMA).
// One workgroup (256 threads = 8 waves) per window; everything resident in
// LDS (177 KB of the 320 KB WGP LDS); single HBM pass (read x, write out).
// f16 WMMA (v_wmma_f32_16x16x32_f16) with fp32 accumulate; flash-attention
// online softmax. rel_bias broadcasts as a per-query-row constant in the
// reference ([1,225,1] vs [B,1,225,225]) and softmax over keys is invariant
// to it, so it is intentionally not applied.
// Round-2 changes: wave-id scalarized via readfirstlane (SALU control flow,
// no exec-mask churn around WMMA), branch-free key masking, float4/v4h
// staging, key loop unrolled x2.
// -------------------------------------------------------------------------

typedef _Float16 f16;
typedef __attribute__((ext_vector_type(16))) _Float16 v16h;
typedef __attribute__((ext_vector_type(8)))  _Float16 v8h;
typedef __attribute__((ext_vector_type(4)))  _Float16 v4h;
typedef __attribute__((ext_vector_type(8)))  float    v8f;

#define N_TOK 225       // tokens per window
#define NP    256       // padded tokens (16 tiles of 16)
#define NQT   15        // query row-tiles actually used (rows < 240)
#define C_DIM 64
#define LDA   72        // padded row stride (f16 elems) for [token][64] bufs
#define LDV   264       // row stride for transposed V [64][NP]
#define LDW   72        // row stride for transposed weights
#define LDP   40        // per-wave P-scratch row stride

// LDS layout (bytes); total 181504 (~177 KB)
#define OFF_X   0                         // 256*72*2 = 36864 (aliased: attn out)
#define OFF_Q   36864
#define OFF_K   73728
#define OFF_VT  110592                    // 64*264*2 = 33792
#define OFF_WT  144384                    // 192*72*2 = 27648 (aliased: P scratch)
#define OFF_WP  172032                    // 64*72*2  = 9216
#define OFF_BI  181248                    // 64*4     = 256
#define LDS_BYTES 181504

static __device__ __forceinline__ v8f wmma16(v16h a, v16h b, v8f c) {
  return __builtin_amdgcn_wmma_f32_16x16x32_f16(false, a, false, b,
                                                (short)0, c, false, false);
}

// A-fragment (16x32 f16): lane holds row (lane%16); lanes<16 carry K {0..7,16..23},
// lanes>=16 carry K {8..15,24..31}.  rowp -> start of this lane's 32-wide K window.
static __device__ __forceinline__ v16h load_a_frag(const f16* rowp, int lane) {
  const int k0 = (lane >> 4) << 3;                 // 0 or 8
  v8h lo = *(const v8h*)(rowp + k0);
  v8h hi = *(const v8h*)(rowp + k0 + 16);
  v16h r;
#pragma unroll
  for (int i = 0; i < 8; ++i) { r[i] = lo[i]; r[i + 8] = hi[i]; }
  return r;
}

// B-fragment (32x16 f16): lane holds column (lane%16); lanes<16 carry K {0..7,16..23},
// lanes>=16 carry K {8..15,24..31}.  colp -> start of this lane's K window.
static __device__ __forceinline__ v16h load_b_frag(const f16* colp, int lane) {
  const int k0 = (lane >> 4) << 3;                 // 0 or 8
  v8h lo = *(const v8h*)(colp + k0);
  v8h hi = *(const v8h*)(colp + k0 + 16);
  v16h r;
#pragma unroll
  for (int i = 0; i < 8; ++i) { r[i] = lo[i]; r[i + 8] = hi[i]; }
  return r;
}

__global__ __launch_bounds__(256)
void winattn_fused_kernel(const float* __restrict__ x,
                          const float* __restrict__ w_qkv,
                          const float* __restrict__ w_proj,
                          const float* __restrict__ b_proj,
                          const float* __restrict__ rel_bias,
                          float* __restrict__ out) {
  (void)rel_bias;  // per-row constant; softmax-invariant (see header comment)

  extern __shared__ char smem[];
  f16*   sX    = (f16*)(smem + OFF_X);   // x (f16), later reused as attn out
  f16*   sQ    = (f16*)(smem + OFF_Q);
  f16*   sK    = (f16*)(smem + OFF_K);
  f16*   sVt   = (f16*)(smem + OFF_VT);  // V transposed: [d][token]
  f16*   sWt   = (f16*)(smem + OFF_WT);  // w_qkv^T, later reused as P scratch
  f16*   sWp   = (f16*)(smem + OFF_WP);  // w_proj^T
  float* sBias = (float*)(smem + OFF_BI);

  const int tid  = threadIdx.x;
  const int lane = tid & 31;
  // wave id is wave-uniform: pin it to an SGPR so tile scheduling, branches
  // and base-address math stay on the SALU (keeps EXEC all-ones at WMMAs).
  const int wv   = __builtin_amdgcn_readfirstlane(tid >> 5);
  const int hi   = lane >> 4;            // half-wave
  const int ln   = lane & 15;
  const int b    = blockIdx.x;
  const size_t xoff = (size_t)b * N_TOK * C_DIM;

  // ---- stage inputs to LDS (f16), zero the padded regions -----------------
  {
    uint4 z{};
    uint4* p0 = (uint4*)sX;
    uint4* p1 = (uint4*)sQ;
    uint4* p2 = (uint4*)sK;
    for (int i = tid; i < NP * LDA * 2 / 16; i += 256) { p0[i] = z; p1[i] = z; p2[i] = z; }
    uint4* p3 = (uint4*)sVt;
    for (int i = tid; i < 64 * LDV * 2 / 16; i += 256) p3[i] = z;
  }
  for (int i = tid; i < N_TOK * C_DIM / 4; i += 256) {       // x: float4 reads
    const float4 v = ((const float4*)(x + xoff))[i];
    const int e = i * 4, r = e >> 6, c = e & 63;
    v4h h; h[0] = (f16)v.x; h[1] = (f16)v.y; h[2] = (f16)v.z; h[3] = (f16)v.w;
    *(v4h*)(sX + r * LDA + c) = h;
  }
  for (int i = tid; i < 64 * 192 / 4; i += 256) {            // w_qkv -> transposed
    const float4 v = ((const float4*)w_qkv)[i];
    const int e = i * 4, k = e / 192, n = e % 192;
    sWt[(n + 0) * LDW + k] = (f16)v.x;
    sWt[(n + 1) * LDW + k] = (f16)v.y;
    sWt[(n + 2) * LDW + k] = (f16)v.z;
    sWt[(n + 3) * LDW + k] = (f16)v.w;
  }
  for (int i = tid; i < 64 * 64 / 4; i += 256) {             // w_proj -> transposed
    const float4 v = ((const float4*)w_proj)[i];
    const int e = i * 4, k = e >> 6, n = e & 63;
    sWp[(n + 0) * LDW + k] = (f16)v.x;
    sWp[(n + 1) * LDW + k] = (f16)v.y;
    sWp[(n + 2) * LDW + k] = (f16)v.z;
    sWp[(n + 3) * LDW + k] = (f16)v.w;
  }
  if (tid < 64) sBias[tid] = b_proj[tid];
  __syncthreads();

  // ---- qkv = x @ w_qkv : [240x64] @ [64x192], 15x12 output tiles ----------
  const float qscale = 0.125f;                               // 1/sqrt(64)
  for (int t = wv; t < NQT * 12; t += 8) {
    const int rt = __builtin_amdgcn_readfirstlane(t / 12);   // scalar tile ids
    const int ct = __builtin_amdgcn_readfirstlane(t - rt * 12);
    const f16* arow = sX + (rt * 16 + ln) * LDA;
    v16h a0 = load_a_frag(arow, lane);
    v16h a1 = load_a_frag(arow + 32, lane);
    const f16* bcol = sWt + (ct * 16 + ln) * LDW;
    v8f acc = {};
    acc = wmma16(a0, load_b_frag(bcol, lane), acc);
    acc = wmma16(a1, load_b_frag(bcol + 32, lane), acc);
    const int rbase = rt * 16 + 8 * hi;
    if (ct < 4) {                                            // Q (pre-scaled)
      const int f = ct * 16 + ln;
#pragma unroll
      for (int j = 0; j < 8; ++j) sQ[(rbase + j) * LDA + f] = (f16)(acc[j] * qscale);
    } else if (ct < 8) {                                     // K
      const int f = (ct - 4) * 16 + ln;
#pragma unroll
      for (int j = 0; j < 8; ++j) sK[(rbase + j) * LDA + f] = (f16)acc[j];
    } else {                                                 // V transposed, packed
      const int f = (ct - 8) * 16 + ln;
      v8h pk;
#pragma unroll
      for (int j = 0; j < 8; ++j) pk[j] = (f16)acc[j];
      *(v8h*)(sVt + f * LDV + rbase) = pk;
    }
  }
  __syncthreads();

  // ---- flash attention + projection, per wave -----------------------------
  f16* sPw = sWt + wv * 16 * LDP;   // 1 KB per-wave P scratch (aliases sWt)
  f16* sA  = sX;                    // attention output (aliases sX)

  for (int qt = wv; qt < NQT; qt += 8) {
    const f16* qrow = sQ + (qt * 16 + ln) * LDA;
    v16h aq0 = load_a_frag(qrow, lane);
    v16h aq1 = load_a_frag(qrow + 32, lane);

    float m[8], l[8];
    v8f o[4];
#pragma unroll
    for (int j = 0; j < 8; ++j) { m[j] = -3.0e38f; l[j] = 0.0f; }
#pragma unroll
    for (int c = 0; c < 4; ++c) o[c] = (v8f){};

#pragma unroll 2
    for (int dk = 0; dk < 8; ++dk) {                         // 32 keys / iter
      const int n0 = dk * 32;
      v8f s0 = {}, s1 = {};
      const f16* k0p = sK + (n0 + ln) * LDA;
      const f16* k1p = sK + (n0 + 16 + ln) * LDA;
      s0 = wmma16(aq0, load_b_frag(k0p, lane), s0);
      s0 = wmma16(aq1, load_b_frag(k0p + 32, lane), s0);
      s1 = wmma16(aq0, load_b_frag(k1p, lane), s1);
      s1 = wmma16(aq1, load_b_frag(k1p + 32, lane), s1);

      // branch-free key masking: add -3e38 on out-of-range key columns
      const float msk0 = (n0 + ln      < N_TOK) ? 0.0f : -3.0e38f;
      const float msk1 = (n0 + 16 + ln < N_TOK) ? 0.0f : -3.0e38f;

#pragma unroll
      for (int j = 0; j < 8; ++j) {
        const float e0 = s0[j] + msk0;
        const float e1 = s1[j] + msk1;
        float v = fmaxf(e0, e1);                             // row max (16 lanes)
        v = fmaxf(v, __shfl_xor(v, 1, 16));
        v = fmaxf(v, __shfl_xor(v, 2, 16));
        v = fmaxf(v, __shfl_xor(v, 4, 16));
        v = fmaxf(v, __shfl_xor(v, 8, 16));
        const float mn    = fmaxf(m[j], v);
        const float alpha = __expf(m[j] - mn);
        const float p0 = __expf(e0 - mn);
        const float p1 = __expf(e1 - mn);
        float r = p0 + p1;                                   // row sum
        r += __shfl_xor(r, 1, 16);
        r += __shfl_xor(r, 2, 16);
        r += __shfl_xor(r, 4, 16);
        r += __shfl_xor(r, 8, 16);
        l[j] = l[j] * alpha + r;
        m[j] = mn;
#pragma unroll
        for (int c = 0; c < 4; ++c) o[c][j] *= alpha;
        sPw[(j + 8 * hi) * LDP + ln]      = (f16)p0;         // C-layout -> scratch
        sPw[(j + 8 * hi) * LDP + 16 + ln] = (f16)p1;
      }

      // P (16x32) back as A-fragment; O += P @ V_tile (4 d-column tiles)
      v16h ap = load_a_frag(sPw + ln * LDP, lane);
#pragma unroll
      for (int c = 0; c < 4; ++c) {
        const f16* vp = sVt + (c * 16 + ln) * LDV + n0;
        o[c] = wmma16(ap, load_b_frag(vp, lane), o[c]);
      }
    }

    // normalize, stash attention output for the projection GEMM
#pragma unroll
    for (int j = 0; j < 8; ++j) {
      const float rl = 1.0f / l[j];
      const int row = qt * 16 + j + 8 * hi;
#pragma unroll
      for (int c = 0; c < 4; ++c) sA[row * LDA + c * 16 + ln] = (f16)(o[c][j] * rl);
    }
  }

  // ---- out = attn_out @ w_proj + b_proj (same-wave rows; no barrier) ------
  for (int qt = wv; qt < NQT; qt += 8) {
    const f16* arow = sA + (qt * 16 + ln) * LDA;
    v16h a0 = load_a_frag(arow, lane);
    v16h a1 = load_a_frag(arow + 32, lane);
#pragma unroll
    for (int ct = 0; ct < 4; ++ct) {
      const f16* bcol = sWp + (ct * 16 + ln) * LDW;
      v8f acc = {};
      acc = wmma16(a0, load_b_frag(bcol, lane), acc);
      acc = wmma16(a1, load_b_frag(bcol + 32, lane), acc);
      const float bb = sBias[ct * 16 + ln];
#pragma unroll
      for (int j = 0; j < 8; ++j) {
        const int row = qt * 16 + j + 8 * hi;
        if (row < N_TOK)
          out[xoff + (size_t)row * C_DIM + ct * 16 + ln] = acc[j] + bb;
      }
    }
  }
}

extern "C" void kernel_launch(void* const* d_in, const int* in_sizes, int n_in,
                              void* d_out, int out_size, void* d_ws, size_t ws_size,
                              hipStream_t stream) {
  (void)in_sizes; (void)n_in; (void)out_size; (void)d_ws; (void)ws_size;
  const float* x        = (const float*)d_in[0];
  const float* w_qkv    = (const float*)d_in[1];
  const float* w_proj   = (const float*)d_in[2];
  const float* b_proj   = (const float*)d_in[3];
  const float* rel_bias = (const float*)d_in[4];
  float* out = (float*)d_out;

  winattn_fused_kernel<<<2048, 256, LDS_BYTES, stream>>>(
      x, w_qkv, w_proj, b_proj, rel_bias, out);
}
